// DemandAwareCrossAttention_62311385531142
// MI455X (gfx1250) — matrix operand order repaired and checked by reading, not compiled, not run on hardware
//
#include <hip/hip_runtime.h>

// ---------------------------------------------------------------------------
// DemandAwareCrossAttention for MI455X (gfx1250, wave32, WMMA)
// All GEMMs use v_wmma_f32_16x16x32_bf16 with f32 accumulation.
// ---------------------------------------------------------------------------

#define P_TOTAL 32768   // H*W = 128*256
#define CDIM    256     // MODEL_DIM

typedef __bf16 bf16;
typedef __bf16 bf16x16 __attribute__((ext_vector_type(16)));
typedef float  f32x8   __attribute__((ext_vector_type(8)));
typedef int    i32x4   __attribute__((ext_vector_type(4)));

union BFrag { i32x4 i[2]; bf16x16 v; };
union StageBuf { unsigned u[8]; i32x4 q[2]; };

// Single-instruction packed f32x2 -> bf16x2 conversion (RNE).
// No builtin documented for gfx1250; the compiler itself emits this opcode,
// so inline asm is safe. Non-volatile: pure value computation.
__device__ __forceinline__ unsigned cvt_pk_bf16x2(float a, float b) {
    unsigned r;
    asm("v_cvt_pk_bf16_f32 %0, %1, %2" : "=v"(r) : "v"(a), "v"(b));
    return r;
}

// ---------------------------------------------------------------------------
// f32 -> bf16 copy (weights)
// ---------------------------------------------------------------------------
__global__ __launch_bounds__(256) void f32_to_bf16_kernel(
    const float* __restrict__ src, bf16* __restrict__ dst, int n)
{
    int i = blockIdx.x * 256 + threadIdx.x;
    if (i < n) dst[i] = (bf16)src[i];
}

// ---------------------------------------------------------------------------
// hidden[h, p] = relu(w_d1[h,:] . demand[:,p] + b_d1[h]) ;  h in [0,128)
// ---------------------------------------------------------------------------
__global__ __launch_bounds__(256) void demand_hidden_kernel(
    const float* __restrict__ demand, const float* __restrict__ w1,
    const float* __restrict__ b1, float* __restrict__ hidden)
{
    const int idx = blockIdx.x * 256 + threadIdx.x;   // 128 * P threads
    const int h = idx >> 15;                          // / P_TOTAL
    const int p = idx & (P_TOTAL - 1);
    float s = b1[h];
    #pragma unroll
    for (int i = 0; i < 3; ++i)
        s += w1[h * 3 + i] * demand[(size_t)i * P_TOTAL + p];
    hidden[idx] = s > 0.f ? s : 0.f;
}

// ---------------------------------------------------------------------------
// Generic WMMA GEMM:  out[b, m, p] = sum_k W[m,k] * X[b,k,p] + bias[m] (+adds)
//   M = 256 (grid.y = 2 tiles of 128), N-tile = 128 pixels, K step = 32.
//   NMAT == 2 computes two weight matrices sharing the same staged B tile
//   (used for the fused K/V projection over the collaborator features).
// ---------------------------------------------------------------------------
template<int KDIM, int NMAT, bool ADD2, bool OUTBF16>
__global__ __launch_bounds__(256) void gemm_bf16_wmma(
    const bf16* __restrict__ W0, const float* __restrict__ bias0, void* __restrict__ out0,
    const bf16* __restrict__ W1, const float* __restrict__ bias1, void* __restrict__ out1,
    const float* __restrict__ X,
    const float* __restrict__ addA, const float* __restrict__ addB)
{
    constexpr int P = P_TOTAL;
    constexpr int KSTEPS = KDIM / 32;
    // per-column record: 32 bf16 (64B) in B-fragment order; stride 80B vs bank conflicts
    __shared__ __align__(16) unsigned char ldsB[128 * 80];

    const int tid  = threadIdx.x;
    const int lane = tid & 31;
    const int wave = tid >> 5;          // 8 waves
    const int lo16 = lane & 15;
    const int hi   = lane >> 4;         // 0 or 1
    const int pbase = blockIdx.x * 128;
    const int m0    = blockIdx.y * 128 + wave * 16;

    const float* Xb = X + (size_t)blockIdx.z * KDIM * P;

    f32x8 acc0[8], acc1[8];
    const f32x8 zero = {0.f, 0.f, 0.f, 0.f, 0.f, 0.f, 0.f, 0.f};
    #pragma unroll
    for (int s = 0; s < 8; ++s) { acc0[s] = zero; acc1[s] = zero; }

    const int col   = tid & 127;        // local pixel column this thread stages
    const int halfB = tid >> 7;         // which 16 K-rows it stages

    for (int kb = 0; kb < KSTEPS; ++kb) {
        __syncthreads();
        {   // stage B tile: f32 global (coalesced) -> bf16 LDS in fragment order
            i32x4* dst = (i32x4*)(ldsB + col * 80 + halfB * 32);
            const int kbase = kb * 32 + halfB * 16;   // linear K 0..15 / 16..31
            float f[16];
            #pragma unroll
            for (int j = 0; j < 16; ++j)
                f[j] = Xb[(size_t)(kbase + j) * P + pbase + col];
            StageBuf sb;
            #pragma unroll
            for (int j = 0; j < 8; ++j)
                sb.u[j] = cvt_pk_bf16x2(f[2 * j], f[2 * j + 1]);
            dst[0] = sb.q[0];
            dst[1] = sb.q[1];
        }
        __syncthreads();

        // A fragment(s): bf16 weights, row-major; contiguous K order (same as B):
        // lane<16 -> K {0..15}, lane>=16 -> K {16..31}
        const int r = m0 + lo16;
        BFrag a0, a1;
        {
            const bf16* ap0 = W0 + (size_t)r * KDIM + kb * 32 + hi * 16;
            a0.i[0] = *(const i32x4*)ap0;
            a0.i[1] = *(const i32x4*)(ap0 + 8);
        }
        if (NMAT == 2) {
            const bf16* ap1 = W1 + (size_t)r * KDIM + kb * 32 + hi * 16;
            a1.i[0] = *(const i32x4*)ap1;
            a1.i[1] = *(const i32x4*)(ap1 + 8);
        }

        #pragma unroll
        for (int s = 0; s < 8; ++s) {
            BFrag b;
            const unsigned char* bp = ldsB + (s * 16 + lo16) * 80 + hi * 32;
            b.i[0] = *(const i32x4*)bp;
            b.i[1] = *(const i32x4*)(bp + 16);
            acc0[s] = __builtin_amdgcn_wmma_f32_16x16x32_bf16(
                false, a0.v, false, b.v, (short)0, acc0[s], false, false);
            if (NMAT == 2)
                acc1[s] = __builtin_amdgcn_wmma_f32_16x16x32_bf16(
                    false, a1.v, false, b.v, (short)0, acc1[s], false, false);
        }
    }

    // Epilogue: D layout -> lane<16: M=j, lane>=16: M=8+j, N=lane&15
    const size_t obase = (size_t)blockIdx.z * CDIM * P;
    #pragma unroll
    for (int s = 0; s < 8; ++s) {
        const int colg = pbase + s * 16 + lo16;
        #pragma unroll
        for (int j = 0; j < 8; ++j) {
            const int m = m0 + hi * 8 + j;
            const size_t oi = obase + (size_t)m * P + colg;
            float v0 = acc0[s][j] + bias0[m];
            if (ADD2) v0 += addA[(size_t)m * P + colg] + addB[(size_t)m * P + colg];
            if (OUTBF16) ((bf16*)out0)[oi] = (bf16)v0;
            else         ((float*)out0)[oi] = v0;
            if (NMAT == 2) {
                float v1 = acc1[s][j] + bias1[m];
                if (OUTBF16) ((bf16*)out1)[oi] = (bf16)v1;
                else         ((float*)out1)[oi] = v1;
            }
        }
    }
}

// ---------------------------------------------------------------------------
// Per-pixel cross attention: softmax over 4 collaborators, 8 heads of dim 32.
// q: bf16 [256,P], k/v: bf16 [4,256,P], out: f32 [256,P]
// ---------------------------------------------------------------------------
__global__ __launch_bounds__(256) void attention_kernel(
    const bf16* __restrict__ q, const bf16* __restrict__ k,
    const bf16* __restrict__ v, float* __restrict__ outp)
{
    const int p = blockIdx.x * 256 + threadIdx.x;
    constexpr int P = P_TOTAL;
    constexpr float scale = 0.17677669529663687f;   // 1/sqrt(32)

    #pragma unroll 1
    for (int m = 0; m < 8; ++m) {
        const int cbase = m * 32;
        float qd[32];
        #pragma unroll
        for (int d = 0; d < 32; ++d)
            qd[d] = (float)q[(size_t)(cbase + d) * P + p];

        float sc[4];
        #pragma unroll
        for (int n = 0; n < 4; ++n) {
            const bf16* kp = k + ((size_t)n * CDIM + cbase) * P + p;
            float s = 0.f;
            #pragma unroll
            for (int d = 0; d < 32; ++d)
                s += qd[d] * (float)kp[(size_t)d * P];
            sc[n] = s * scale;
        }
        const float mx = fmaxf(fmaxf(sc[0], sc[1]), fmaxf(sc[2], sc[3]));
        float e[4];
        float sum = 0.f;
        #pragma unroll
        for (int n = 0; n < 4; ++n) { e[n] = __expf(sc[n] - mx); sum += e[n]; }
        const float inv = 1.f / sum;

        float od[32];
        #pragma unroll
        for (int d = 0; d < 32; ++d) od[d] = 0.f;
        #pragma unroll
        for (int n = 0; n < 4; ++n) {
            const float wn = e[n] * inv;
            const bf16* vp = v + ((size_t)n * CDIM + cbase) * P + p;
            #pragma unroll
            for (int d = 0; d < 32; ++d)
                od[d] += wn * (float)vp[(size_t)d * P];
        }
        #pragma unroll
        for (int d = 0; d < 32; ++d)
            outp[(size_t)(cbase + d) * P + p] = od[d];
    }
}

// ---------------------------------------------------------------------------
// Host-side launcher
// ---------------------------------------------------------------------------
extern "C" void kernel_launch(void* const* d_in, const int* in_sizes, int n_in,
                              void* d_out, int out_size, void* d_ws, size_t ws_size,
                              hipStream_t stream)
{
    const float* ego    = (const float*)d_in[0];
    const float* demand = (const float*)d_in[1];
    const float* collab = (const float*)d_in[2];
    const float* w_d1   = (const float*)d_in[3];
    const float* b_d1   = (const float*)d_in[4];
    const float* w_d2   = (const float*)d_in[5];
    const float* b_d2   = (const float*)d_in[6];
    const float* wq     = (const float*)d_in[7];
    const float* bq     = (const float*)d_in[8];
    const float* wk     = (const float*)d_in[9];
    const float* bk     = (const float*)d_in[10];
    const float* wv     = (const float*)d_in[11];
    const float* bv     = (const float*)d_in[12];
    const float* wo     = (const float*)d_in[13];
    const float* bo     = (const float*)d_in[14];
    const float* pos    = (const float*)d_in[15];

    const size_t P = P_TOTAL;
    char* ws = (char*)d_ws;
    size_t off = 0;
    auto carve = [&](size_t bytes) -> void* {
        void* p = ws + off;
        off += (bytes + 255) & ~(size_t)255;
        return p;
    };
    bf16*  wq_b   = (bf16*)carve(65536 * 2);
    bf16*  wk_b   = (bf16*)carve(65536 * 2);
    bf16*  wv_b   = (bf16*)carve(65536 * 2);
    bf16*  wo_b   = (bf16*)carve(65536 * 2);
    bf16*  wd2_b  = (bf16*)carve(32768 * 2);
    float* hidden = (float*)carve(128 * P * 4);
    float* qstate = (float*)carve(256 * P * 4);
    bf16*  q_b    = (bf16*)carve(256 * P * 2);
    bf16*  k_b    = (bf16*)carve(4 * 256 * P * 2);
    bf16*  v_b    = (bf16*)carve(4 * 256 * P * 2);
    float* attn   = (float*)carve(256 * P * 4);

    // 1) weights -> bf16
    f32_to_bf16_kernel<<<256, 256, 0, stream>>>(wq, wq_b, 65536);
    f32_to_bf16_kernel<<<256, 256, 0, stream>>>(wk, wk_b, 65536);
    f32_to_bf16_kernel<<<256, 256, 0, stream>>>(wv, wv_b, 65536);
    f32_to_bf16_kernel<<<256, 256, 0, stream>>>(wo, wo_b, 65536);
    f32_to_bf16_kernel<<<128, 256, 0, stream>>>(w_d2, wd2_b, 32768);

    // 2) demand MLP hidden layer (3 -> 128, relu)
    demand_hidden_kernel<<<(128 * P_TOTAL) / 256, 256, 0, stream>>>(demand, w_d1, b_d1, hidden);

    // 3) qstate = W_d2 . hidden + b_d2 + ego + pos          (K=128, f32 out)
    gemm_bf16_wmma<128, 1, true, false><<<dim3(P_TOTAL / 128, 2, 1), 256, 0, stream>>>(
        wd2_b, b_d2, qstate, nullptr, nullptr, nullptr, hidden, ego, pos);

    // 4) q = Wq . qstate + bq                               (bf16 out)
    gemm_bf16_wmma<256, 1, false, true><<<dim3(P_TOTAL / 128, 2, 1), 256, 0, stream>>>(
        wq_b, bq, q_b, nullptr, nullptr, nullptr, qstate, nullptr, nullptr);

    // 5) fused k/v = Wk/Wv . collab + bk/bv over 4 batches  (bf16 out)
    gemm_bf16_wmma<256, 2, false, true><<<dim3(P_TOTAL / 128, 2, 4), 256, 0, stream>>>(
        wk_b, bk, k_b, wv_b, bv, v_b, collab, nullptr, nullptr);

    // 6) per-pixel softmax attention over collaborators
    attention_kernel<<<P_TOTAL / 256, 256, 0, stream>>>(q_b, k_b, v_b, attn);

    // 7) out = Wo . attn + bo                               (f32 out -> d_out)
    gemm_bf16_wmma<256, 1, false, false><<<dim3(P_TOTAL / 128, 2, 1), 256, 0, stream>>>(
        wo_b, bo, d_out, nullptr, nullptr, nullptr, attn, nullptr, nullptr);
}
